// Head_29575144800535
// MI455X (gfx1250) — matrix operand order repaired and checked by reading, not compiled, not run on hardware
//
#include <hip/hip_runtime.h>
#include <hip/hip_bf16.h>

typedef __attribute__((ext_vector_type(16))) __bf16 v16bf;
typedef __attribute__((ext_vector_type(8)))  __bf16 v8bf;
typedef __attribute__((ext_vector_type(8)))  float  v8f;
typedef __attribute__((ext_vector_type(4)))  unsigned int v4u;
typedef __attribute__((ext_vector_type(8)))  unsigned int v8u;

#define B_  32
#define T_  1024
#define C_  256
#define HS_ 64

static __device__ __forceinline__ unsigned short f2bf(float f) {
  return __builtin_bit_cast(unsigned short, (__bf16)f);
}

static __device__ __forceinline__ v8f wmma_bf16(v16bf a, v16bf b, v8f c) {
  // D = A(16x32) * B(32x16) + C, f32 accumulate
  return __builtin_amdgcn_wmma_f32_16x16x32_bf16(false, a, false, b, (short)0, c, false, false);
}

static __device__ __forceinline__ float rmax16(float v) {
  v = fmaxf(v, __shfl_xor(v, 1, 32));
  v = fmaxf(v, __shfl_xor(v, 2, 32));
  v = fmaxf(v, __shfl_xor(v, 4, 32));
  v = fmaxf(v, __shfl_xor(v, 8, 32));
  return v;
}
static __device__ __forceinline__ float rsum16(float v) {
  v += __shfl_xor(v, 1, 32);
  v += __shfl_xor(v, 2, 32);
  v += __shfl_xor(v, 4, 32);
  v += __shfl_xor(v, 8, 32);
  return v;
}

// truncate generic LDS pointer -> wave LDS byte offset (ISA 10.2: addr[31:0])
static __device__ __forceinline__ unsigned int lds_off(const void* p) {
  return (unsigned int)(uintptr_t)p;
}

// A-fragment (16x32 bf16): per lane 16 contiguous bf16 at +hig*16
static __device__ __forceinline__ v16bf ld_afrag(const unsigned short* p, int hig) {
  const v8bf lo = *(const v8bf*)(p + hig * 16);
  const v8bf hi = *(const v8bf*)(p + hig * 16 + 8);
  v16bf r;
  #pragma unroll
  for (int i = 0; i < 8; ++i) { r[i] = lo[i]; r[8 + i] = hi[i]; }
  return r;
}

// ---------------------------------------------------------------------------
// TDM: DMA a 2D tile (tile_w x tile_h bf16, row stride in elements) to LDS.
// D# built per ISA 8.3/8.4: group0 {count=1, lds_addr, global_addr, type=2},
// group1 {data_size=2B, dims, tile dims, dim0 stride}; groups 2/3 zero (2D).
// Tracked by TENSORcnt; EXEC ignored.
// ---------------------------------------------------------------------------
static __device__ __forceinline__ void tdm_load_2d(
    unsigned int lds_byte, const void* gaddr,
    unsigned int tile_w, unsigned int tile_h, unsigned long long stride_elems)
{
  const unsigned long long ga = (unsigned long long)(uintptr_t)gaddr;
  v4u g0;
  g0[0] = 1u;                                                    // 1 valid descriptor
  g0[1] = lds_byte;                                              // lds_addr
  g0[2] = (unsigned int)ga;                                      // global_addr[31:0]
  g0[3] = (unsigned int)((ga >> 32) & 0x1ffffffu) | (2u << 30);  // addr[56:32], type=2
  v8u g1;
  g1[0] = 1u << 16;                                              // data_size=2B
  g1[1] = (tile_w & 0xffffu) << 16;                              // tensor_dim0[15:0]
  g1[2] = ((tile_w >> 16) & 0xffffu) | ((tile_h & 0xffffu) << 16);
  g1[3] = ((tile_h >> 16) & 0xffffu) | ((tile_w & 0xffffu) << 16); // + tile_dim0
  g1[4] = tile_h & 0xffffu;                                      // tile_dim1, dim2=0
  g1[5] = (unsigned int)stride_elems;                            // dim0_stride[31:0]
  g1[6] = (unsigned int)((stride_elems >> 32) & 0xffffu);        // dim0_stride[47:32]
  g1[7] = 0u;
  const v4u gz = (v4u){0u, 0u, 0u, 0u};
  asm volatile("tensor_load_to_lds %0, %1, %2, %3"
               :: "s"(g0), "s"(g1), "s"(gz), "s"(gz) : "memory");
}

// ---------------------------------------------------------------------------
// Kernel 0: one-shot weight conversion f32 -> bf16 (Wq pre-scaled by 1/sqrt(HS))
// ---------------------------------------------------------------------------
__global__ __launch_bounds__(256, 1) void convert_w(
    const float* __restrict__ Wk, const float* __restrict__ Wq,
    const float* __restrict__ Wv,
    unsigned short* __restrict__ wkb, unsigned short* __restrict__ wqb,
    unsigned short* __restrict__ wvb)
{
  const int i = blockIdx.x * 256 + threadIdx.x;   // 64 blocks -> 16384 = HS*C
  wkb[i] = f2bf(Wk[i]);
  wqb[i] = f2bf(Wq[i] * 0.125f);
  wvb[i] = f2bf(Wv[i]);
}

// ---------------------------------------------------------------------------
// Kernel 1: fused Q/K/V projection, x[32768,256]f32 @ bf16-W^T -> bf16.
// Weights staged once per block in LDS via 3 TDM loads (96KB dynamic LDS).
// V stored transposed [B,HS,T]. One wave = one 16-row tile.
// ---------------------------------------------------------------------------
__global__ __launch_bounds__(256, 1) void proj_qkv(
    const float* __restrict__ x,
    const unsigned short* __restrict__ wkb, const unsigned short* __restrict__ wqb,
    const unsigned short* __restrict__ wvb,
    unsigned short* __restrict__ qb, unsigned short* __restrict__ kb,
    unsigned short* __restrict__ vtb)
{
  extern __shared__ __align__(32) unsigned short wlds[];  // [q|k|v] 3*16384 bf16
  const int lane = threadIdx.x & 31;
  const int wave = threadIdx.x >> 5;
  const int row0 = (blockIdx.x * 8 + wave) << 4;
  const int lrow = lane & 15;
  const int hig  = (lane >> 4) & 1;

  if (wave == 0) {
    tdm_load_2d(lds_off(&wlds[0]),         wqb, C_, HS_, C_);
    tdm_load_2d(lds_off(&wlds[HS_ * C_]),  wkb, C_, HS_, C_);
    tdm_load_2d(lds_off(&wlds[2*HS_*C_]),  wvb, C_, HS_, C_);
  }
  __builtin_amdgcn_s_wait_tensorcnt((unsigned short)0);
  __syncthreads();

  v8f aq[4], ak[4], av[4];
  #pragma unroll
  for (int n = 0; n < 4; ++n) {
    aq[n] = (v8f){0,0,0,0,0,0,0,0};
    ak[n] = (v8f){0,0,0,0,0,0,0,0};
    av[n] = (v8f){0,0,0,0,0,0,0,0};
  }

  for (int ks = 0; ks < C_; ks += 32) {
    // A fragment from x (f32 -> bf16): 16 contiguous floats per lane at +hig*16
    const float* xr = x + (size_t)(row0 + lrow) * C_ + ks;
    const float4 c0 = *(const float4*)(xr + hig * 16);
    const float4 c1 = *(const float4*)(xr + hig * 16 + 4);
    const float4 c2 = *(const float4*)(xr + hig * 16 + 8);
    const float4 c3 = *(const float4*)(xr + hig * 16 + 12);
    v16bf a;
    a[0]=(__bf16)c0.x;  a[1]=(__bf16)c0.y;  a[2]=(__bf16)c0.z;  a[3]=(__bf16)c0.w;
    a[4]=(__bf16)c1.x;  a[5]=(__bf16)c1.y;  a[6]=(__bf16)c1.z;  a[7]=(__bf16)c1.w;
    a[8]=(__bf16)c2.x;  a[9]=(__bf16)c2.y;  a[10]=(__bf16)c2.z; a[11]=(__bf16)c2.w;
    a[12]=(__bf16)c3.x; a[13]=(__bf16)c3.y; a[14]=(__bf16)c3.z; a[15]=(__bf16)c3.w;

    #pragma unroll
    for (int n = 0; n < 4; ++n) {
      // B fragment: 16 contiguous bf16 per lane from LDS weights
      const int off = (n*16 + lrow) * C_ + ks + hig * 16;
      const v16bf bq = *(const v16bf*)&wlds[off];
      const v16bf bk = *(const v16bf*)&wlds[HS_*C_ + off];
      const v16bf bv = *(const v16bf*)&wlds[2*HS_*C_ + off];
      aq[n] = wmma_bf16(a, bq, aq[n]);
      ak[n] = wmma_bf16(a, bk, ak[n]);
      av[n] = wmma_bf16(a, bv, av[n]);
    }
  }

  const int b  = row0 >> 10;
  const int t0 = row0 & (T_ - 1);
  #pragma unroll
  for (int n = 0; n < 4; ++n) {
    #pragma unroll
    for (int g = 0; g < 8; ++g) {
      const int M = g + hig * 8;
      const int h = n * 16 + lrow;
      qb[(size_t)(row0 + M) * HS_ + h] = f2bf(aq[n][g]);   // scale folded into Wq
      kb[(size_t)(row0 + M) * HS_ + h] = f2bf(ak[n][g]);
      vtb[((size_t)b * HS_ + h) * T_ + (t0 + M)] = f2bf(av[n][g]);
    }
  }
}

// ---------------------------------------------------------------------------
// Kernel 2: causal flash attention. Block = 8 waves x 16 query rows = 128 rows.
// K/V 32-key tiles staged once per block via TDM (wave 0 issues, TENSORcnt),
// double buffered; waves past their causal range skip compute uniformly.
// ---------------------------------------------------------------------------
__global__ __launch_bounds__(256, 1) void attn_fwd(
    const unsigned short* __restrict__ qb, const unsigned short* __restrict__ kb,
    const unsigned short* __restrict__ vtb, float* __restrict__ out)
{
  __shared__ __align__(32) unsigned short ktile[2][32 * HS_]; // [key][h]   8KB
  __shared__ __align__(32) unsigned short vtile[2][HS_ * 32]; // [h][key]   8KB
  __shared__ __align__(32) unsigned short ptile[8 * 16 * 32]; // per-wave P 8KB

  const int lane  = threadIdx.x & 31;
  const int wave  = threadIdx.x >> 5;
  const int b     = blockIdx.x >> 3;
  const int qbase = (blockIdx.x & 7) * 128;
  const int qrow0 = qbase + wave * 16;
  const int lrow  = lane & 15;
  const int hig   = (lane >> 4) & 1;

  unsigned short* my = ptile + wave * 512;

  // ---- Q A-fragments (head dim 64 = 2 K-steps), pre-scaled by 1/sqrt(HS) ----
  v16bf qa[2];
  const unsigned short* qp = qb + (size_t)(b * T_ + qrow0 + lrow) * HS_;
  qa[0] = ld_afrag(qp, hig);
  qa[1] = ld_afrag(qp + 32, hig);

  v8f acc[4];
  #pragma unroll
  for (int n = 0; n < 4; ++n) acc[n] = (v8f){0,0,0,0,0,0,0,0};
  float m[8], l[8];
  #pragma unroll
  for (int g = 0; g < 8; ++g) { m[g] = -1e30f; l[g] = 0.0f; }

  // per 32-key tile: K tile 32x64 (stride 64), V^T tile 64x32 (stride T)
  #define ISSUE_TILE(KT, BUF)                                                   \
    do {                                                                        \
      tdm_load_2d(lds_off(&ktile[(BUF)][0]),                                    \
                  kb + (size_t)(b * T_ + (KT) * 32) * HS_, HS_, 32, HS_);       \
      tdm_load_2d(lds_off(&vtile[(BUF)][0]),                                    \
                  vtb + (size_t)b * HS_ * T_ + (KT) * 32, 32, HS_, T_);         \
    } while (0)

  const int ktiles = (qbase + 128) / 32;   // union causal range for the block
  if (wave == 0) ISSUE_TILE(0, 0);

  for (int kt = 0; kt < ktiles; ++kt) {
    const int kt32 = kt * 32;
    const int buf  = kt & 1;

    if (wave == 0) {
      if (kt + 1 < ktiles) {
        ISSUE_TILE(kt + 1, buf ^ 1);
        __builtin_amdgcn_s_wait_tensorcnt((unsigned short)2); // tile kt done (in-order)
      } else {
        __builtin_amdgcn_s_wait_tensorcnt((unsigned short)0);
      }
    }
    __syncthreads();                                          // tile kt visible

    if (kt32 < qrow0 + 16) {   // wave-uniform causal skip; EXEC stays all-ones
      // ---- S = Q K^T : two 16x16 D tiles; B-frags contiguous in LDS K tile ----
      v8f d0 = (v8f){0,0,0,0,0,0,0,0};
      v8f d1 = (v8f){0,0,0,0,0,0,0,0};
      #pragma unroll
      for (int f = 0; f < 2; ++f) {
        const int hoff = f * 32 + hig * 16;
        const v16bf b0 = *(const v16bf*)&ktile[buf][lrow * HS_ + hoff];
        const v16bf b1 = *(const v16bf*)&ktile[buf][(lrow + 16) * HS_ + hoff];
        d0 = wmma_bf16(qa[f], b0, d0);
        d1 = wmma_bf16(qa[f], b1, d1);
      }

      // ---- causal mask + online softmax (row stats across 16-lane groups) ----
      float scv[8], e0[8], e1[8];
      #pragma unroll
      for (int g = 0; g < 8; ++g) {
        const int row = qrow0 + g + hig * 8;
        const int c0  = kt32 + lrow;
        const float s0 = (c0      <= row) ? d0[g] : -1e30f;
        const float s1 = (c0 + 16 <= row) ? d1[g] : -1e30f;
        const float tm = rmax16(fmaxf(s0, s1));
        const float mn = fmaxf(m[g], tm);
        const float sc = __expf(m[g] - mn);
        const float x0 = __expf(s0 - mn);
        const float x1 = __expf(s1 - mn);
        l[g] = l[g] * sc + rsum16(x0 + x1);
        m[g] = mn;
        scv[g] = sc; e0[g] = x0; e1[g] = x1;
      }
      #pragma unroll
      for (int n = 0; n < 4; ++n)
        #pragma unroll
        for (int g = 0; g < 8; ++g)
          acc[n][g] *= scv[g];

      // ---- P: D-layout -> per-wave LDS (16x32 bf16) -> A-fragment ----
      #pragma unroll
      for (int g = 0; g < 8; ++g) {
        const int M = g + hig * 8;
        my[M*32 + lrow]      = f2bf(e0[g]);
        my[M*32 + 16 + lrow] = f2bf(e1[g]);
      }
      __builtin_amdgcn_wave_barrier();
      asm volatile("s_wait_dscnt 0x0" ::: "memory");  // LDS in-order per wave
      const v16bf pa = ld_afrag(my + lrow * 32, hig);
      __builtin_amdgcn_wave_barrier();

      // ---- O += P V : B-frags contiguous in LDS V tile ([h][key]) ----
      #pragma unroll
      for (int n = 0; n < 4; ++n) {
        const v16bf vb = *(const v16bf*)&vtile[buf][(n*16 + lrow) * 32 + hig * 16];
        acc[n] = wmma_bf16(pa, vb, acc[n]);
      }
    }

    __syncthreads();   // all waves done reading buf before tile kt+2 overwrites it
  }
  #undef ISSUE_TILE

  // ---- epilogue: normalize and store f32 ----
  float inv[8];
  #pragma unroll
  for (int g = 0; g < 8; ++g) inv[g] = 1.0f / l[g];
  #pragma unroll
  for (int n = 0; n < 4; ++n)
    #pragma unroll
    for (int g = 0; g < 8; ++g) {
      const int M = g + hig * 8;
      out[((size_t)(b * T_ + qrow0 + M)) * HS_ + n*16 + lrow] = acc[n][g] * inv[g];
    }
}

// ---------------------------------------------------------------------------
extern "C" void kernel_launch(void* const* d_in, const int* in_sizes, int n_in,
                              void* d_out, int out_size, void* d_ws, size_t ws_size,
                              hipStream_t stream) {
  const float* x  = (const float*)d_in[0];
  const float* Wk = (const float*)d_in[1];
  const float* Wq = (const float*)d_in[2];
  const float* Wv = (const float*)d_in[3];
  float* out = (float*)d_out;

  // workspace: bf16 Q [32768,64], K [32768,64], V^T [32,64,1024], W's [64,256]x3
  unsigned short* qbuf = (unsigned short*)d_ws;
  unsigned short* kbuf = qbuf + (size_t)B_ * T_ * HS_;
  unsigned short* vtb  = kbuf + (size_t)B_ * T_ * HS_;
  unsigned short* wkb  = vtb  + (size_t)B_ * T_ * HS_;
  unsigned short* wqb  = wkb  + (size_t)HS_ * C_;
  unsigned short* wvb  = wqb  + (size_t)HS_ * C_;

  convert_w<<<dim3(64),  dim3(256), 0, stream>>>(Wk, Wq, Wv, wkb, wqb, wvb);
  proj_qkv <<<dim3(256), dim3(256), 3 * HS_ * C_ * sizeof(unsigned short), stream>>>(
      x, wkb, wqb, wvb, qbuf, kbuf, vtb);
  attn_fwd <<<dim3(256), dim3(256), 0, stream>>>(qbuf, kbuf, vtb, out);
}
